// DynamicMasksLoss_w_Matching_35107062677774
// MI455X (gfx1250) — matrix operand order repaired and checked by reading, not compiled, not run on hardware
//
#include <hip/hip_runtime.h>
#include <hip/hip_bf16.h>

// Problem constants (match reference)
#define B_ 8
#define Q_ 25
#define H_ 256
#define W_ 256
#define HW_ (H_*W_)
#define D_ 256
#define L_ 20
#define EPS_ 1e-4f
#define SMOOTH_ 1.0f

typedef __attribute__((ext_vector_type(16))) _Float16 v16h;
typedef __attribute__((ext_vector_type(8)))  float    v8f;

// ---------------- workspace slot map (floats) ----------------
#define S_CE     0            // 1: sum of logp[b,gt,hw]
#define S_INTER  1            // 16: [b*2+c] sum sm*onehot
#define S_SMSUM  17           // 16: [b*2+c] sum sm
#define S_CNT1   33           // 8:  [b] count of merged_gt==1 (== sum m)
#define S_MASKS  48           // 200*9: per (b,q) mask-loss partial sums
#define S_CONTR  1856         // 8: sum over l of (1 - picked)
#define S_ABSG   1864         // 8: sum |G| over 25x25
#define S_DIAG   1872         // 8: sum diag(G)
#define S_SIZE   1880         // 8: sum max(1 - sizes, 0)
#define WS_ZERO  2048

// ---------------- reductions (wave32) ----------------
__device__ __forceinline__ float waveReduce(float v) {
#pragma unroll
  for (int o = 16; o > 0; o >>= 1) v += __shfl_down(v, o, 32);
  return v;
}
__device__ __forceinline__ float blockReduce(float v, float* sh) {
  int lane = threadIdx.x & 31, wid = threadIdx.x >> 5;
  v = waveReduce(v);
  if (lane == 0) sh[wid] = v;
  __syncthreads();
  int nw = blockDim.x >> 5;
  v = (threadIdx.x < nw) ? sh[threadIdx.x] : 0.0f;
  if (wid == 0) v = waveReduce(v);
  return v;  // valid on thread 0
}

// ---------------- 0: zero the atomic workspace ----------------
__global__ void init_kernel(float* __restrict__ ws) {
  int i = blockIdx.x * blockDim.x + threadIdx.x;
  if (i < WS_ZERO) ws[i] = 0.0f;
}

// ---------------- 1: merged CE + dice partials ----------------
// grid = B*64 blocks, 256 thr, 4 px/thr (float4 / int4 b128 loads)
__global__ __launch_bounds__(256) void merged_kernel(const float* __restrict__ mp,
                                                     const int* __restrict__ mg,
                                                     float* __restrict__ ws) {
  __shared__ float red[8];
  int blk = blockIdx.x;
  int b = blk >> 6, s = blk & 63;
  int off = s * 1024 + threadIdx.x * 4;
  float4 x0 = *(const float4*)(mp + (size_t)b * 2 * HW_ + off);
  float4 x1 = *(const float4*)(mp + (size_t)b * 2 * HW_ + HW_ + off);
  int4   g4 = *(const int4*)(mg + (size_t)b * HW_ + off);
  float a0[4] = {x0.x, x0.y, x0.z, x0.w};
  float a1[4] = {x1.x, x1.y, x1.z, x1.w};
  int   gs[4] = {g4.x, g4.y, g4.z, g4.w};

  float ce = 0, in0 = 0, in1 = 0, s0 = 0, s1 = 0, c1 = 0;
#pragma unroll
  for (int u = 0; u < 4; ++u) {
    float mx = fmaxf(a0[u], a1[u]);
    float e0 = __expf(a0[u] - mx), e1 = __expf(a1[u] - mx);
    float inv = 1.0f / (e0 + e1);
    float lse = mx + __logf(e0 + e1);
    float lp0 = a0[u] - lse, lp1 = a1[u] - lse;
    float p0 = e0 * inv, p1 = e1 * inv;
    int g = gs[u];
    ce += g ? lp1 : lp0;
    in0 += (g == 0) ? p0 : 0.0f;
    in1 += (g == 1) ? p1 : 0.0f;
    s0 += p0; s1 += p1;
    c1 += (float)g;
  }
  float vals[6] = {ce, in0, in1, s0, s1, c1};
  int slots[6] = {S_CE, S_INTER + b * 2 + 0, S_INTER + b * 2 + 1,
                  S_SMSUM + b * 2 + 0, S_SMSUM + b * 2 + 1, S_CNT1 + b};
  for (int k = 0; k < 6; ++k) {
    float v = blockReduce(vals[k], red);
    if (threadIdx.x == 0) atomicAdd(&ws[slots[k]], v);
    __syncthreads();
  }
}

// ---------------- 2: per-(b,q) mask-loss partial sums ----------------
// grid = B*Q*8 blocks; each block reduces 8192 elems (256 thr * 8 * float4)
__global__ __launch_bounds__(256) void masks_kernel(const float* __restrict__ mpred,
                                                    const float* __restrict__ mgt,
                                                    const int* __restrict__ merged_gt,
                                                    float* __restrict__ ws) {
  __shared__ float red[8];
  int blk = blockIdx.x;
  int s = blk & 7;
  int bq = blk >> 3;
  int b = bq / Q_;
  size_t base  = (size_t)bq * HW_ + s * 8192;
  size_t mbase = (size_t)b * HW_ + s * 8192;

  float acc[9];
#pragma unroll
  for (int k = 0; k < 9; ++k) acc[k] = 0.0f;

  for (int it = 0; it < 8; ++it) {
    int off = it * 1024 + threadIdx.x * 4;
    float4 x4 = *(const float4*)(mpred + base + off);
    float4 t4 = *(const float4*)(mgt + base + off);
    int4   g4 = *(const int4*)(merged_gt + mbase + off);
    float xs[4] = {x4.x, x4.y, x4.z, x4.w};
    float ts[4] = {t4.x, t4.y, t4.z, t4.w};
    int   gs[4] = {g4.x, g4.y, g4.z, g4.w};
#pragma unroll
    for (int u = 0; u < 4; ++u) {
      float p   = 1.0f / (1.0f + __expf(-xs[u]));
      float l1m = __logf(1.0f - p + EPS_);
      float lpp = __logf(p + EPS_);
      float t = ts[u];
      float m = (float)gs[u];
      float a = (1.0f - t) * m;           // (1-t)*m
      float c = (1.0f - t) * (1.0f - m);  // (1-t)*(1-m)
      acc[0] += p * t;                    // inter_q
      acc[1] += p;                        // sum p
      acc[2] += t;                        // sum t
      acc[3] += -l1m * a;                 // numer lf1
      acc[4] += a;                        // denom lf1
      acc[5] += -l1m * c;                 // numer lf2
      acc[6] += c;                        // denom lf2
      acc[7] += -lpp * t;                 // numer lf3
      acc[8] += -l1m * m;                 // numer lbg
    }
  }
  for (int k = 0; k < 9; ++k) {
    float v = blockReduce(acc[k], red);
    if (threadIdx.x == 0) atomicAdd(&ws[S_MASKS + bq * 9 + k], v);
    __syncthreads();
  }
}

// ---------------- 3: matching loss via WMMA (f16 in / f32 acc) ----------------
// One workgroup per batch; 4 waves; each wave owns one 16x16 tile of the
// 32x32-padded output. Pass 0: S = (query @ text^T) * D^-0.5 ; Pass 1: G = query @ query^T.
//
// Zero-padded f16 operands are staged in LDS first (unconditional float4
// global loads with clamped row index + VALU select for the pad rows), so the
// WMMA fragment-gather loop is branch-free: no EXEC masking, no per-load
// s_wait_loadcnt, and the K-loop fully unrolls into back-to-back v_wmma.
//
// A fragment (16x32 f16), lane (hi=lane>>4, M=lane&15): half-pair j<4 holds
// K=8*hi+2j{,+1}; j>=4 holds K=16+8*hi+2(j-4){,+1}. B fragment (32x16 f16):
// lane N=lane&15 holds K=8*hi+i (i<8), K=16+8*hi+(i-8) (i>=8), same as A.
__global__ __launch_bounds__(128) void matching_kernel(const float* __restrict__ query,
                                                       const float* __restrict__ text,
                                                       const int* __restrict__ text_label,
                                                       float* __restrict__ ws) {
  __shared__ _Float16 Aq[32][D_];   // padded query rows (f16), 16 KB
  __shared__ _Float16 Bt[32][D_];   // padded text rows (f16), 16 KB
  __shared__ float Smat[32][33];    // one 32x32 result (padded stride)

  int b = blockIdx.x;
  int tid = threadIdx.x;
  const float* qbase = query + (size_t)b * Q_ * D_;
  const float* tbase = text + (size_t)b * L_ * D_;

  // ---- stage padded f16 operands in LDS (unconditional vector loads) ----
  for (int idx = tid; idx < 32 * (D_ / 4); idx += 128) {
    int row = idx >> 6;
    int col = (idx & 63) * 4;
    float4 vq = *(const float4*)(qbase + (size_t)(row < Q_ ? row : 0) * D_ + col);
    float4 vt = *(const float4*)(tbase + (size_t)(row < L_ ? row : 0) * D_ + col);
    bool qr = row < Q_, tr = row < L_;
    Aq[row][col + 0] = (_Float16)(qr ? vq.x : 0.0f);
    Aq[row][col + 1] = (_Float16)(qr ? vq.y : 0.0f);
    Aq[row][col + 2] = (_Float16)(qr ? vq.z : 0.0f);
    Aq[row][col + 3] = (_Float16)(qr ? vq.w : 0.0f);
    Bt[row][col + 0] = (_Float16)(tr ? vt.x : 0.0f);
    Bt[row][col + 1] = (_Float16)(tr ? vt.y : 0.0f);
    Bt[row][col + 2] = (_Float16)(tr ? vt.z : 0.0f);
    Bt[row][col + 3] = (_Float16)(tr ? vt.w : 0.0f);
  }
  __syncthreads();

  int wave = tid >> 5, lane = tid & 31;
  int mi = wave & 1, ni = wave >> 1;
  int hi = lane >> 4;
  int mrow = (lane & 15) + 16 * mi;
  int ncol = (lane & 15) + 16 * ni;

  for (int pass = 0; pass < 2; ++pass) {
    const _Float16(*Bm)[D_] = pass ? Aq : Bt;

    v8f acc = {};
#pragma unroll
    for (int kb = 0; kb < D_; kb += 32) {
      v16h a, bf;
#pragma unroll
      for (int i = 0; i < 16; ++i) {
        int j = i >> 1;
        int k = ((j < 4) ? (8 * hi + 2 * j) : (16 + 8 * hi + 2 * (j - 4))) + (i & 1);
        a[i] = Aq[mrow][kb + k];
      }
#pragma unroll
      for (int i = 0; i < 16; ++i)
        bf[i] = Bm[ncol][kb + 8 * hi + i + ((i >= 8) ? 8 : 0)];
      acc = __builtin_amdgcn_wmma_f32_16x16x32_f16(false, a, false, bf,
                                                   (short)0, acc, false, false);
    }
    float scale = pass ? 1.0f : 0.0625f;  // D^-0.5 = 1/sqrt(256)
#pragma unroll
    for (int r = 0; r < 8; ++r)
      Smat[16 * mi + 8 * hi + r][ncol] = acc[r] * scale;
    __syncthreads();

    if (pass == 0) {
      if (tid < L_) {
        int l = tid;
        int lab = text_label[b * L_ + l];
        int idx = ((lab == -1) ? Q_ : lab) - 1;  // in [0, Q-1]
        float mx = -1e30f;
        for (int q = 0; q < Q_; ++q) mx = fmaxf(mx, Smat[q][l]);
        float se = 0.0f;
        for (int q = 0; q < Q_; ++q) se += __expf(Smat[q][l] - mx);
        float picked = __expf(Smat[idx][l] - mx) / se;
        atomicAdd(&ws[S_CONTR + b], 1.0f - picked);
      }
    } else {
      if (tid < Q_) {
        int q = tid;
        float sa = 0.0f;
        for (int k = 0; k < Q_; ++k) sa += fabsf(Smat[q][k]);
        float diag = Smat[q][q];                 // = sizes[q] = ||query_q||^2
        atomicAdd(&ws[S_ABSG + b], sa);
        atomicAdd(&ws[S_DIAG + b], fabsf(diag));
        atomicAdd(&ws[S_SIZE + b], fmaxf(1.0f - diag, 0.0f));
      }
    }
    __syncthreads();
  }
}

// ---------------- 4: finalize (tiny scalar work) ----------------
__global__ void finalize_kernel(const float* __restrict__ ws,
                                const float* __restrict__ cq_pred,
                                const int* __restrict__ cq_gt,
                                const float* __restrict__ obn_pred,
                                const int* __restrict__ obn_gt,
                                float* __restrict__ out) {
  if (threadIdx.x != 0 || blockIdx.x != 0) return;

  // merged: CE + dice
  float ce = -ws[S_CE] / (float)(B_ * HW_);
  float dsum = 0.0f;
  for (int b = 0; b < B_; ++b) {
    float cnt1 = ws[S_CNT1 + b];
    for (int c = 0; c < 2; ++c) {
      float inter = ws[S_INTER + b * 2 + c];
      float sms = ws[S_SMSUM + b * 2 + c];
      float ohs = c ? cnt1 : ((float)HW_ - cnt1);
      dsum += 1.0f - (2.0f * inter + SMOOTH_) / (sms + ohs + SMOOTH_);
    }
  }
  float loss_merged = ce + dsum / (float)(B_ * 2);

  // masks
  float lm_sum = 0.0f;
  for (int b = 0; b < B_; ++b) {
    float nfg = (float)obn_gt[b];
    float cnt1 = ws[S_CNT1 + b];
    float f1 = 0, f2 = 0, f3 = 0, dc = 0, bg = 0;
    for (int q = 0; q < Q_; ++q) {
      const float* s = ws + S_MASKS + (b * Q_ + q) * 9;
      float dice_q = 1.0f - (2.0f * s[0] + SMOOTH_) / (s[1] + s[2] + SMOOTH_);
      float lf1 = s[3] / (s[4] + 1.0f);
      float lf2 = s[5] / (s[6] + 1.0f);
      float lf3 = s[7] / (s[2] + 1.0f);
      float lbg = s[8] / (cnt1 + 1.0f);
      if (q < obn_gt[b]) { dc += dice_q; f1 += lf1; f2 += lf2; f3 += lf3; }
      else               { bg += lbg; }
    }
    f1 /= nfg; f2 /= nfg; f3 /= nfg; dc /= nfg;
    float fg_loss = fmaxf(fmaxf(f1, f3), f2);
    float bg_loss = bg / ((float)Q_ - nfg);
    lm_sum += 0.5f * fg_loss + 0.2f * bg_loss + 0.3f * dc;
  }
  float loss_masks = lm_sum / (float)B_;

  // class CE over first obnum queries
  float lc_sum = 0.0f;
  for (int b = 0; b < B_; ++b) {
    float nfg = (float)obn_gt[b];
    float acc = 0.0f;
    for (int q = 0; q < obn_gt[b]; ++q) {
      float a0 = cq_pred[(b * Q_ + q) * 2 + 0];
      float a1 = cq_pred[(b * Q_ + q) * 2 + 1];
      float mx = fmaxf(a0, a1);
      float lse = mx + __logf(__expf(a0 - mx) + __expf(a1 - mx));
      int g = cq_gt[b * Q_ + q];
      acc += -((g ? a1 : a0) - lse);
    }
    lc_sum += acc / nfg;
  }
  float loss_class = lc_sum / (float)B_;

  // smooth-L1
  float ln_sum = 0.0f;
  for (int b = 0; b < B_; ++b) {
    float x = obn_pred[b] - (float)obn_gt[b] / (float)Q_;
    float ax = fabsf(x);
    ln_sum += (ax < 1.0f) ? 0.5f * x * x : (ax - 0.5f);
  }
  float loss_num = ln_sum / (float)B_;

  // matching
  float contr = 0.0f, orth = 0.0f, szl = 0.0f;
  for (int b = 0; b < B_; ++b) {
    contr += ws[S_CONTR + b];
    float ob = (ws[S_ABSG + b] - ws[S_DIAG + b]) * 0.5f;
    orth += ob / ((float)Q_ * (float)(Q_ - 1) * 0.5f);
    szl += ws[S_SIZE + b] / (float)(Q_ - 1);
  }
  contr /= (float)(B_ * L_);
  orth /= (float)B_;
  szl  /= (float)B_;
  float loss_matching = contr + 0.5f * szl + 0.5f * orth;

  float total = 0.2f * loss_merged + 0.6f * loss_masks + 0.1f * loss_class
              + 0.1f * loss_num + 0.05f * loss_matching;
  out[0] = total;
  out[1] = loss_merged;
  out[2] = loss_masks;
  out[3] = loss_class;
  out[4] = loss_num;
  out[5] = loss_matching;
}

extern "C" void kernel_launch(void* const* d_in, const int* in_sizes, int n_in,
                              void* d_out, int out_size, void* d_ws, size_t ws_size,
                              hipStream_t stream) {
  const float* merged_pred = (const float*)d_in[0];
  const int*   merged_gt   = (const int*)d_in[1];
  const float* masks_pred  = (const float*)d_in[2];
  const float* masks_gt    = (const float*)d_in[3];
  const float* cquery_pred = (const float*)d_in[4];
  const int*   cquery_gt   = (const int*)d_in[5];
  const float* obnum_pred  = (const float*)d_in[6];
  const int*   obnum_gt    = (const int*)d_in[7];
  const float* query       = (const float*)d_in[8];
  const float* text        = (const float*)d_in[9];
  const int*   text_label  = (const int*)d_in[10];
  float* ws  = (float*)d_ws;
  float* out = (float*)d_out;

  hipLaunchKernelGGL(init_kernel, dim3(WS_ZERO / 256), dim3(256), 0, stream, ws);
  hipLaunchKernelGGL(merged_kernel, dim3(B_ * 64), dim3(256), 0, stream,
                     merged_pred, merged_gt, ws);
  hipLaunchKernelGGL(masks_kernel, dim3(B_ * Q_ * 8), dim3(256), 0, stream,
                     masks_pred, masks_gt, merged_gt, ws);
  hipLaunchKernelGGL(matching_kernel, dim3(B_), dim3(128), 0, stream,
                     query, text, text_label, ws);
  hipLaunchKernelGGL(finalize_kernel, dim3(1), dim3(32), 0, stream,
                     ws, cquery_pred, cquery_gt, obnum_pred, obnum_gt, out);
}
